// CustomStaticEdgeConv_996432413183
// MI455X (gfx1250) — compile-verified, matrix operand order checked by reading, not run on hardware
//
#include <hip/hip_runtime.h>
#include <hip/hip_bf16.h>

typedef __attribute__((ext_vector_type(16))) _Float16 v16h;
typedef __attribute__((ext_vector_type(8)))  _Float16 v8h;
typedef __attribute__((ext_vector_type(4)))  _Float16 v4h;
typedef __attribute__((ext_vector_type(8)))  float    v8f;

#define F_IN   64
#define F_OUT  128
#define KDIM   128            // 2*F_IN
#define RS     136            // padded LDS row stride in halves (128 + 8)

// ---------------- init: zero output accumulator + degree counts ----------------
__global__ void ec_init_kernel(float* __restrict__ out, float* __restrict__ counts,
                               int n_nodes) {
    int t = blockIdx.x * blockDim.x + threadIdx.x;
    int total = n_nodes * F_OUT;
    if (t < total)   out[t] = 0.0f;
    if (t < n_nodes) counts[t] = 0.0f;
}

// ---------------- main edge kernel: gather -> f16 WMMA -> ReLU -> atomic scatter ----------------
__global__ __launch_bounds__(256)
void ec_edge_kernel(const float* __restrict__ x,
                    const int*   __restrict__ ei,     // (2, E): [0..E) = row, [E..2E) = col
                    const float* __restrict__ W,      // (F_OUT, KDIM) row-major
                    const float* __restrict__ bias,   // (F_OUT)
                    float* __restrict__ out,          // (N, F_OUT) accumulator
                    float* __restrict__ counts,       // (N)
                    int n_edges, int ntiles)
{
    __shared__ _Float16 Wlds[F_OUT * RS];       // 34816 B, W as f16 (B matrix source)
    __shared__ _Float16 Ebuf[8 * 16 * RS];      // 34816 B, 16 edges x 128 feats per wave

    const int tid  = threadIdx.x;
    const int wave = tid >> 5;
    const int lane = tid & 31;
    const int r16  = lane & 15;                 // M/N index within a 16-wide tile
    const int hs   = lane >> 4;                 // which half this lane holds

    // ---- stage W (128x128 f32) -> LDS f16, padded row-major (once per block) ----
    const float4* W4 = (const float4*)W;
    #pragma unroll
    for (int it = 0; it < 16; ++it) {
        int idx4 = tid + it * 256;              // 4096 float4 total
        int n = idx4 >> 5;                      // 32 float4 per 128-wide row
        int k = (idx4 & 31) << 2;
        float4 wv = W4[idx4];
        v4h h; h[0] = (_Float16)wv.x; h[1] = (_Float16)wv.y;
               h[2] = (_Float16)wv.z; h[3] = (_Float16)wv.w;
        *(v4h*)&Wlds[n * RS + k] = h;
    }

    // ---- hoist bias values this lane needs (tile-invariant) ----
    float bvs[8];
    #pragma unroll
    for (int nb = 0; nb < 8; ++nb) bvs[nb] = bias[nb * 16 + r16];

    __syncthreads();

    const int el = lane >> 1;                   // edge within wave tile (2 lanes/edge)
    const int hh = lane & 1;                    // half: 0 = x_r, 1 = x_c - x_r
    _Float16* stage_dst = &Ebuf[(wave * 16 + el) * RS + hh * F_IN];
    const _Float16* arow = &Ebuf[(wave * 16 + r16) * RS];

    // ---- persistent loop over 128-edge tiles ----
    for (int tile = blockIdx.x; tile < ntiles; tile += gridDim.x) {
        const int edgebase = tile * 128 + wave * 16;

        // stage this wave's 16 edges: [x_r | x_c - x_r] as f16
        {
            int m = edgebase + el;
            bool valid = (m < n_edges);
            int r = valid ? ei[m] : 0;
            int c = valid ? ei[n_edges + m] : 0;
            const float4* xr = (const float4*)(x + (size_t)r * F_IN);
            const float4* xc = (const float4*)(x + (size_t)c * F_IN);
            #pragma unroll
            for (int k4 = 0; k4 < 16; ++k4) {
                float4 a = xr[k4];
                if (hh) {
                    float4 bb = xc[k4];
                    a.x = bb.x - a.x; a.y = bb.y - a.y; a.z = bb.z - a.z; a.w = bb.w - a.w;
                }
                if (!valid) { a.x = 0.f; a.y = 0.f; a.z = 0.f; a.w = 0.f; }
                v4h hv; hv[0] = (_Float16)a.x; hv[1] = (_Float16)a.y;
                        hv[2] = (_Float16)a.z; hv[3] = (_Float16)a.w;
                *(v4h*)&stage_dst[k4 * 4] = hv;
            }
        }

        // degree counts: one lane per edge
        if (lane < 16) {
            int mg = edgebase + lane;
            if (mg < n_edges) unsafeAtomicAdd(&counts[ei[mg]], 1.0f);
        }

        __syncthreads();   // make staged LDS visible across lanes

        // A fragments (16 edges x K=32 per chunk), ISA 16-bit A layout
        v16h afrag[4];
        #pragma unroll
        for (int kc = 0; kc < 4; ++kc) {
            union { v16h v; struct { v8h lo, hi; } s; } u;
            u.s.lo = *(const v8h*)&arow[kc * 32 + hs * 8];
            u.s.hi = *(const v8h*)&arow[kc * 32 + 16 + hs * 8];
            afrag[kc] = u.v;
        }

        // per-lane destination pointers + validity masks (M = hs*8 + j)
        float* aptr[8];
        float  vmask[8];
        #pragma unroll
        for (int j = 0; j < 8; ++j) {
            int mg = edgebase + hs * 8 + j;
            bool mv = (mg < n_edges);
            int rw = mv ? ei[mg] : 0;
            aptr[j]  = out + (size_t)rw * F_OUT + r16;
            vmask[j] = mv ? 1.0f : 0.0f;
        }

        // 8 N-tiles x 4 K-chunks of v_wmma_f32_16x16x32_f16 (fully unrolled:
        // constant nb -> bias in fixed regs, nb*16 folds into atomic ioffset)
        #pragma unroll
        for (int nb = 0; nb < 8; ++nb) {
            const _Float16* brow = &Wlds[(nb * 16 + r16) * RS];   // B col n = W row n
            v8f acc = {};
            #pragma unroll
            for (int kc = 0; kc < 4; ++kc) {
                union { v16h v; struct { v8h lo, hi; } s; } u;
                u.s.lo = *(const v8h*)&brow[kc * 32 + hs * 8];
                u.s.hi = *(const v8h*)&brow[kc * 32 + 16 + hs * 8];
                acc = __builtin_amdgcn_wmma_f32_16x16x32_f16(
                          false, afrag[kc], false, u.v, (short)0, acc, false, false);
            }
            #pragma unroll
            for (int j = 0; j < 8; ++j) {
                float v = acc[j] + bvs[nb];
                v = (v > 0.0f ? v : 0.0f) * vmask[j];   // ReLU + tail mask (no divergence)
                unsafeAtomicAdd(aptr[j] + nb * 16, v);  // constant offset -> ioffset
            }
        }
        // no end-of-iteration barrier needed: next staging only conflicts
        // with this wave's own in-order DS accesses
    }
}

// ---------------- finalize: mean-normalize by clamped degree ----------------
__global__ void ec_final_kernel(float* __restrict__ out, const float* __restrict__ counts,
                                int n_nodes) {
    int t = blockIdx.x * blockDim.x + threadIdx.x;
    int total = n_nodes * F_OUT;
    if (t < total) {
        float c = counts[t / F_OUT];
        out[t] = out[t] / fmaxf(c, 1.0f);
    }
}

extern "C" void kernel_launch(void* const* d_in, const int* in_sizes, int n_in,
                              void* d_out, int out_size, void* d_ws, size_t ws_size,
                              hipStream_t stream) {
    const float* x  = (const float*)d_in[0];
    const int*   ei = (const int*)d_in[1];
    const float* W  = (const float*)d_in[2];
    const float* b  = (const float*)d_in[3];
    float* out    = (float*)d_out;
    float* counts = (float*)d_ws;               // n_nodes floats of scratch

    const int n_nodes = in_sizes[0] / F_IN;
    const int n_edges = in_sizes[1] / 2;
    const int total   = n_nodes * F_OUT;

    ec_init_kernel<<<(total + 255) / 256, 256, 0, stream>>>(out, counts, n_nodes);

    const int ntiles = (n_edges + 127) / 128;   // 128 edges per 256-thread block
    int blocks = ntiles < 2048 ? ntiles : 2048; // persistent blocks; W staged once each
    ec_edge_kernel<<<blocks, 256, 0, stream>>>(x, ei, W, b, out, counts, n_edges, ntiles);

    ec_final_kernel<<<(total + 255) / 256, 256, 0, stream>>>(out, counts, n_nodes);
}
